// CGCNNBlock_65420941853353
// MI455X (gfx1250) — matrix operand order repaired and checked by reading, not compiled, not run on hardware
//
#include <hip/hip_runtime.h>

#define N_NODES 50000
#define N_EDGES 800000
#define ATOM 128
#define EDGE_F 64
#define HID 128
#define KCAT 192 /* ATOM + EDGE_F */

typedef __attribute__((ext_vector_type(16))) __bf16 v16bf;
typedef __attribute__((ext_vector_type(8))) float v8f;
typedef __attribute__((ext_vector_type(4))) float f32x4;
typedef __attribute__((ext_vector_type(4))) unsigned int u32x4;

union FragB16 {
  v16bf v;
  u32x4 q[2];
  __bf16 b[16];
};

__device__ __forceinline__ float silu_f(float v) {
  return v * (1.0f / (1.0f + __expf(-v)));
}

// Async global->LDS 16B copy (GVS mode: SGPR base + VGPR byte offset).
// Tracked with ASYNCcnt; fence with s_wait_asynccnt before use.
__device__ __forceinline__ void async_copy16(unsigned lds_off,
                                             const void* gbase, int goff) {
  asm volatile("global_load_async_to_lds_b128 %0, %1, %2 offset:0"
               :
               : "v"(lds_off), "v"(goff), "s"(gbase)
               : "memory");
}

__device__ __forceinline__ void async_fence() {
  asm volatile("s_wait_asynccnt 0x0" ::: "memory");
}

// Convert 16 f32 (two 8-float runs) to a 16x32 bf16 A-fragment lane slice.
__device__ __forceinline__ void cvtA(FragB16& a, const f32x4* r) {
#pragma unroll
  for (int t = 0; t < 4; ++t) {
    a.b[t]      = (__bf16)r[0][t];
    a.b[4 + t]  = (__bf16)r[1][t];
    a.b[8 + t]  = (__bf16)r[2][t];
    a.b[12 + t] = (__bf16)r[3][t];
  }
}

__device__ __forceinline__ void loadB(FragB16& b, const __bf16* base, int nt,
                                      int lrow, int kcol, int ldk) {
  const __bf16* bp = base + (nt * 16 + lrow) * ldk + kcol;
  b.q[0] = *(const u32x4*)(bp);
  b.q[1] = *(const u32x4*)(bp + 16);
}

// ---------------------------------------------------------------------------
// One-time: convert W1/W2/W3 (f32, K x N row-major) to bf16 transposed (N x K)
// in workspace:  Wt1[128][192] @ 0, Wt2[128][128] @ 24576, Wt3[128][128] @ 40960
// ---------------------------------------------------------------------------
__global__ void prep_weights(const float* __restrict__ W1,
                             const float* __restrict__ W2,
                             const float* __restrict__ W3,
                             __bf16* __restrict__ wt) {
  int tid = blockIdx.x * blockDim.x + threadIdx.x;
  if (tid < HID * KCAT) {
    int n = tid / KCAT, k = tid % KCAT;
    wt[tid] = (__bf16)W1[k * HID + n];
  } else if (tid < HID * KCAT + HID * HID) {
    int t = tid - HID * KCAT;
    int n = t / HID, k = t % HID;
    wt[HID * KCAT + n * HID + k] = (__bf16)W2[k * HID + n];
  } else if (tid < HID * KCAT + 2 * HID * HID) {
    int t = tid - HID * KCAT - HID * HID;
    int n = t / HID, k = t % HID;
    wt[HID * KCAT + HID * HID + n * HID + k] = (__bf16)W3[k * ATOM + n];
  }
}

// Zero the aggregation table (d_out). Grid sized exactly: N*HID/4 float4.
__global__ void zero_f32(float* __restrict__ p) {
  long i = (long)blockIdx.x * blockDim.x + threadIdx.x;
  f32x4 z = {0.0f, 0.0f, 0.0f, 0.0f};
  ((f32x4*)p)[i] = z;
}

// ---------------------------------------------------------------------------
// Edge MLP: 32 edges per wave (two 16-row A tiles share every B fragment),
// 256 edges per block. 160 WMMAs per wave.
// ---------------------------------------------------------------------------
__global__ void __launch_bounds__(256)
edge_mlp(const float* __restrict__ x, const int* __restrict__ edge_i,
         const int* __restrict__ edge_j, const float* __restrict__ edge_f,
         const __bf16* __restrict__ wt, const float* __restrict__ b1,
         const float* __restrict__ b2, float* __restrict__ agg) {
  extern __shared__ __bf16 smem[];
  __bf16* sW1 = smem;                          // 24576 elems (48 KB)
  __bf16* sW2 = smem + HID * KCAT;             // 16384 elems (32 KB)
  __bf16* sH  = smem + HID * KCAT + HID * HID; // 8 waves * 32*128 (64 KB)

  // Stage bf16 weights (Wt1 + Wt2 contiguous in ws) into LDS via async DMA.
  {
    const unsigned lds0 = (unsigned)(size_t)(void*)smem; // low 32b = LDS offset
    const int total = (HID * KCAT + HID * HID) / 8;      // 5120 x 16B
    for (int i = threadIdx.x; i < total; i += 256)
      async_copy16(lds0 + i * 16, wt, i * 16);
  }
  async_fence();
  __syncthreads();

  const int wave = threadIdx.x >> 5;
  const int lane = threadIdx.x & 31;
  const int lrow = lane & 15;
  const int hi   = lane >> 4;
  const long ebase = (long)blockIdx.x * 256 + wave * 32;
  const long e0 = ebase + lrow;
  const long e1 = ebase + 16 + lrow;

  const int j0 = edge_j[e0];
  const int j1 = edge_j[e1];
  const float* rowx0 = x + (long)j0 * ATOM;
  const float* rowx1 = x + (long)j1 * ATOM;
  const float* rowe0 = edge_f + e0 * EDGE_F;
  const float* rowe1 = edge_f + e1 * EDGE_F;

  v8f acc0[8], acc1[8];
#pragma unroll
  for (int t = 0; t < 8; ++t)
#pragma unroll
    for (int r = 0; r < 8; ++r) { acc0[t][r] = 0.0f; acc1[t][r] = 0.0f; }

  // raw f32 A loads for one K-step (two 8-float runs per tile)
  auto ldraw = [&](int kb, f32x4* r, const float* rowx, const float* rowe) {
    const int kbase = kb * 32 + hi * 8;
    const int k16 = kbase + 16;
    const float* s0 = (kbase < ATOM) ? (rowx + kbase) : (rowe + kbase - ATOM);
    const float* s1 = (k16 < ATOM) ? (rowx + k16) : (rowe + k16 - ATOM);
    r[0] = ((const f32x4*)s0)[0];
    r[1] = ((const f32x4*)s0)[1];
    r[2] = ((const f32x4*)s1)[0];
    r[3] = ((const f32x4*)s1)[1];
  };

  // ---- Layer 1: K = 192, 6 K-steps of 32, pipelined A (global) + B (LDS) ----
  f32x4 r0[4], r1[4];
  ldraw(0, r0, rowx0, rowe0);
  ldraw(0, r1, rowx1, rowe1);
#pragma unroll
  for (int kb = 0; kb < 6; ++kb) {
    FragB16 a0, a1;
    cvtA(a0, r0);
    cvtA(a1, r1);
    if (kb < 5) {
      ldraw(kb + 1, r0, rowx0, rowe0);
      ldraw(kb + 1, r1, rowx1, rowe1);
    }
    const int kcol = kb * 32 + hi * 8;
    FragB16 bc, bn;
    loadB(bc, sW1, 0, lrow, kcol, KCAT);
#pragma unroll
    for (int nt = 0; nt < 8; ++nt) {
      if (nt < 7) loadB(bn, sW1, nt + 1, lrow, kcol, KCAT);
      acc0[nt] = __builtin_amdgcn_wmma_f32_16x16x32_bf16(
          false, a0.v, false, bc.v, (short)0, acc0[nt], false, false);
      acc1[nt] = __builtin_amdgcn_wmma_f32_16x16x32_bf16(
          false, a1.v, false, bc.v, (short)0, acc1[nt], false, false);
      if (nt < 7) bc = bn;
    }
  }

  // bias + SiLU, re-layout D->A via per-wave LDS tile (32 x 128 bf16)
  __bf16* sHw = sH + wave * (32 * HID);
#pragma unroll
  for (int nt = 0; nt < 8; ++nt) {
    const int n = nt * 16 + lrow;
    const float bias = b1[n];
#pragma unroll
    for (int r = 0; r < 8; ++r) {
      const int m = r + hi * 8;
      sHw[m * HID + n]        = (__bf16)silu_f(acc0[nt][r] + bias);
      sHw[(16 + m) * HID + n] = (__bf16)silu_f(acc1[nt][r] + bias);
      acc0[nt][r] = 0.0f;
      acc1[nt][r] = 0.0f;
    }
  }
  asm volatile("s_wait_dscnt 0x0" ::: "memory");
  __builtin_amdgcn_wave_barrier();

  // ---- Layer 2: K = 128, 4 K-steps of 32, pipelined A + B (both LDS) ----
  auto ldA2 = [&](int kb, FragB16& a, int mrow) {
    const int kbase = kb * 32 + hi * 8;
    const __bf16* ap = sHw + mrow * HID + kbase;
    a.q[0] = *(const u32x4*)(ap);
    a.q[1] = *(const u32x4*)(ap + 16);
  };
  FragB16 a0, a1, a0n, a1n;
  ldA2(0, a0, lrow);
  ldA2(0, a1, 16 + lrow);
#pragma unroll
  for (int kb = 0; kb < 4; ++kb) {
    if (kb < 3) {
      ldA2(kb + 1, a0n, lrow);
      ldA2(kb + 1, a1n, 16 + lrow);
    }
    const int kcol = kb * 32 + hi * 8;
    FragB16 bc, bn;
    loadB(bc, sW2, 0, lrow, kcol, HID);
#pragma unroll
    for (int nt = 0; nt < 8; ++nt) {
      if (nt < 7) loadB(bn, sW2, nt + 1, lrow, kcol, HID);
      acc0[nt] = __builtin_amdgcn_wmma_f32_16x16x32_bf16(
          false, a0.v, false, bc.v, (short)0, acc0[nt], false, false);
      acc1[nt] = __builtin_amdgcn_wmma_f32_16x16x32_bf16(
          false, a1.v, false, bc.v, (short)0, acc1[nt], false, false);
      if (nt < 7) bc = bn;
    }
    if (kb < 3) { a0 = a0n; a1 = a1n; }
  }

  // bias + SiLU + scatter-add into agg
  int dsts0[8], dsts1[8];
#pragma unroll
  for (int r = 0; r < 8; ++r) {
    dsts0[r] = edge_i[ebase + r + hi * 8];
    dsts1[r] = edge_i[ebase + 16 + r + hi * 8];
  }
#pragma unroll
  for (int nt = 0; nt < 8; ++nt) {
    const int n = nt * 16 + lrow;
    const float bias = b2[n];
#pragma unroll
    for (int r = 0; r < 8; ++r) {
      const float v0 = silu_f(acc0[nt][r] + bias);
      const float v1 = silu_f(acc1[nt][r] + bias);
      __hip_atomic_fetch_add(&agg[(long)dsts0[r] * HID + n], v0,
                             __ATOMIC_RELAXED, __HIP_MEMORY_SCOPE_AGENT);
      __hip_atomic_fetch_add(&agg[(long)dsts1[r] * HID + n], v1,
                             __ATOMIC_RELAXED, __HIP_MEMORY_SCOPE_AGENT);
    }
  }
}

// ---------------------------------------------------------------------------
// Node projection: out = silu(x + agg @ W3 + b3), in-place over agg (d_out).
// 16 rows per wave, 128 rows per block. 32 WMMAs per wave.
// ---------------------------------------------------------------------------
__global__ void __launch_bounds__(256)
node_proj(const float* __restrict__ x, const __bf16* __restrict__ wt3,
          const float* __restrict__ b3, float* __restrict__ out) {
  __shared__ __bf16 sW3[HID * HID];
  {
    const unsigned lds0 = (unsigned)(size_t)(void*)sW3;
    for (int i = threadIdx.x; i < HID * HID / 8; i += 256)
      async_copy16(lds0 + i * 16, wt3, i * 16);
  }
  async_fence();
  __syncthreads();

  const int wave = threadIdx.x >> 5;
  const int lane = threadIdx.x & 31;
  const int lrow = lane & 15;
  const int hi   = lane >> 4;
  const long rbase = (long)blockIdx.x * 128 + wave * 16;
  long row = rbase + lrow;
  if (row > N_NODES - 1) row = N_NODES - 1; // clamp for tail-block loads
  const float* aggrow = out + row * HID;

  v8f acc[8];
#pragma unroll
  for (int t = 0; t < 8; ++t)
#pragma unroll
    for (int r = 0; r < 8; ++r) acc[t][r] = 0.0f;

  f32x4 rr[4];
  rr[0] = ((const f32x4*)(aggrow + hi * 8))[0];
  rr[1] = ((const f32x4*)(aggrow + hi * 8))[1];
  rr[2] = ((const f32x4*)(aggrow + hi * 8 + 16))[0];
  rr[3] = ((const f32x4*)(aggrow + hi * 8 + 16))[1];
#pragma unroll
  for (int kb = 0; kb < 4; ++kb) {
    FragB16 a;
    cvtA(a, rr);
    if (kb < 3) {
      const int kn = (kb + 1) * 32 + hi * 8;
      rr[0] = ((const f32x4*)(aggrow + kn))[0];
      rr[1] = ((const f32x4*)(aggrow + kn))[1];
      rr[2] = ((const f32x4*)(aggrow + kn + 16))[0];
      rr[3] = ((const f32x4*)(aggrow + kn + 16))[1];
    }
    const int kcol = kb * 32 + hi * 8;
    FragB16 bc, bn;
    loadB(bc, sW3, 0, lrow, kcol, HID);
#pragma unroll
    for (int nt = 0; nt < 8; ++nt) {
      if (nt < 7) loadB(bn, sW3, nt + 1, lrow, kcol, HID);
      acc[nt] = __builtin_amdgcn_wmma_f32_16x16x32_bf16(
          false, a.v, false, bc.v, (short)0, acc[nt], false, false);
      if (nt < 7) bc = bn;
    }
  }

  // residual + bias + SiLU, in-place store
#pragma unroll
  for (int nt = 0; nt < 8; ++nt) {
    const int n = nt * 16 + lrow;
    const float bias = b3[n];
#pragma unroll
    for (int r = 0; r < 8; ++r) {
      const long row2 = rbase + r + hi * 8;
      if (row2 < N_NODES) {
        const float v = x[row2 * HID + n] + acc[nt][r] + bias;
        out[row2 * HID + n] = silu_f(v);
      }
    }
  }
}

extern "C" void kernel_launch(void* const* d_in, const int* in_sizes, int n_in,
                              void* d_out, int out_size, void* d_ws, size_t ws_size,
                              hipStream_t stream) {
  const float* x      = (const float*)d_in[0];
  const int*   edge_i = (const int*)d_in[1];
  const int*   edge_j = (const int*)d_in[2];
  const float* edge_f = (const float*)d_in[3];
  const float* W1 = (const float*)d_in[4];
  const float* b1 = (const float*)d_in[5];
  const float* W2 = (const float*)d_in[6];
  const float* b2 = (const float*)d_in[7];
  const float* W3 = (const float*)d_in[8];
  const float* b3 = (const float*)d_in[9];
  float* out = (float*)d_out;
  __bf16* wt = (__bf16*)d_ws;

  // 1) bf16/transposed weights into workspace
  const int welems = HID * KCAT + 2 * HID * HID; // 57344
  prep_weights<<<(welems + 255) / 256, 256, 0, stream>>>(W1, W2, W3, wt);

  // 2) zero the aggregation table (d_out doubles as agg buffer)
  zero_f32<<<(N_NODES * HID / 4) / 256, 256, 0, stream>>>(out); // 6250 blocks exact

  // 3) edge MLP + scatter (256 edges per block)
  const size_t lds1 = (size_t)(HID * KCAT + HID * HID + 8 * 32 * HID) *
                      sizeof(__bf16); // 147456 B
  edge_mlp<<<N_EDGES / 256, 256, lds1, stream>>>(x, edge_i, edge_j, edge_f, wt,
                                                 b1, b2, out);

  // 4) node projection + residual + SiLU (in-place on d_out)
  node_proj<<<(N_NODES + 127) / 128, 256, 0, stream>>>(
      x, wt + HID * KCAT + HID * HID, b3, out);
}
